// MultiQueryAttention_84713934946428
// MI455X (gfx1250) — matrix operand order repaired and checked by reading, not compiled, not run on hardware
//
#include <hip/hip_runtime.h>
#include <hip/hip_bf16.h>

typedef __bf16 bf16_t;
typedef __attribute__((ext_vector_type(16))) __bf16 v16bf;
typedef __attribute__((ext_vector_type(8)))  __bf16 v8bf;
typedef __attribute__((ext_vector_type(8)))  float   v8f;

#define BATCH 2
#define SEQ   2048
#define EMB   1024
#define NH    16
#define HD    64

static __device__ __forceinline__ v16bf cat8(v8bf a, v8bf b) {
  return __builtin_shufflevector(a, b, 0,1,2,3,4,5,6,7,8,9,10,11,12,13,14,15);
}

static __device__ __forceinline__ v8f zero8() {
  v8f z;
#pragma unroll
  for (int i = 0; i < 8; ++i) z[i] = 0.0f;
  return z;
}

// A-fragment: 16x32 bf16 tile at `base` (row-major, row stride ld).
// gfx1250 layout: lane L -> row L%16; lo lanes hold K = kk+{0..7,16..23},
// hi lanes hold K = kk+{8..15,24..31}.
static __device__ __forceinline__ v16bf load_a(const bf16_t* base, int ld, int kk, int lane) {
  const bf16_t* p = base + (size_t)(lane & 15) * ld + kk + ((lane >= 16) ? 8 : 0);
  v8bf c0 = *(const v8bf*)p;
  v8bf c1 = *(const v8bf*)(p + 16);
  return cat8(c0, c1);
}

// B-fragment: 32x16 bf16 tile, source stored column-contiguous:
// element B[k][n] at base[n*ld + k]. lane L -> col L%16; lo lanes K=kk+{0..7,16..23},
// hi lanes K=kk+{8..15,24..31}.
static __device__ __forceinline__ v16bf load_b_cm(const bf16_t* base, int ld, int kk, int lane) {
  const bf16_t* p = base + (size_t)(lane & 15) * ld + kk + ((lane >= 16) ? 8 : 0);
  v8bf c0 = *(const v8bf*)p;
  v8bf c1 = *(const v8bf*)(p + 16);
  return cat8(c0, c1);
}

static __device__ __forceinline__ v8f wmma_bf16(v16bf a, v16bf b, v8f c) {
  return __builtin_amdgcn_wmma_f32_16x16x32_bf16(false, a, false, b, (short)0, c, false, false);
}

// ---------------------------------------------------------------- conversions
__global__ void cvt_f32_bf16(const float* __restrict__ in, bf16_t* __restrict__ out, int n) {
  int i = blockIdx.x * blockDim.x + threadIdx.x;
  if (i < n) out[i] = (bf16_t)in[i];
}

// in [R][C] fp32 -> out [C][R] bf16
__global__ void transpose_f32_bf16(const float* __restrict__ in, bf16_t* __restrict__ out,
                                   int R, int C) {
  int i = blockIdx.x * blockDim.x + threadIdx.x;
  if (i < R * C) {
    int r = i / C, c = i % C;
    out[(size_t)c * R + r] = (bf16_t)in[i];
  }
}

// ---------------------------------------------------------------- GEMM (skinny N)
// out[M][N] = A[M][K](bf16) x Wt[N][K](bf16) + bias. One 16x16 tile per wave.
__global__ __launch_bounds__(256)
void gemm_bf16_16x16(const bf16_t* __restrict__ A, const bf16_t* __restrict__ Wt,
                     const float* __restrict__ bias, float* __restrict__ outf,
                     bf16_t* __restrict__ outh, int M, int N, int K) {
  const int lane = threadIdx.x & 31;
  const int wave = threadIdx.x >> 5;
  const int l15  = lane & 15;
  const bool hi  = lane >= 16;
  const int m0 = blockIdx.y * 32 + (wave & 1) * 16;
  const int n0 = blockIdx.x * 64 + (wave >> 1) * 16;

  const bf16_t* abase = A  + (size_t)m0 * K;
  const bf16_t* bbase = Wt + (size_t)n0 * K;

  v8f acc = zero8();
#pragma unroll 4
  for (int kk = 0; kk < K; kk += 32) {
    acc = wmma_bf16(load_a(abase, K, kk, lane), load_b_cm(bbase, K, kk, lane), acc);
  }

  const int col = n0 + l15;
  const float bb = bias[col];
#pragma unroll
  for (int r = 0; r < 8; ++r) {
    const int row = m0 + r + (hi ? 8 : 0);
    const float v = acc[r] + bb;
    if (outf) outf[(size_t)row * N + col] = v;
    if (outh) outh[(size_t)row * N + col] = (bf16_t)v;
  }
}

// ---------------------------------------------------------------- GEMM (wide N)
// 2x2 of 16x16 tiles per wave (32x32): each fragment reused twice ->
// 4 wmma per 4 fragment loads. Block = 8 waves as 2(M) x 4(N) = 64 x 128 tile.
__global__ __launch_bounds__(256)
void gemm_bf16_32x32(const bf16_t* __restrict__ A, const bf16_t* __restrict__ Wt,
                     const float* __restrict__ bias, float* __restrict__ outf,
                     bf16_t* __restrict__ outh, int M, int N, int K) {
  const int lane = threadIdx.x & 31;
  const int wave = threadIdx.x >> 5;
  const int l15  = lane & 15;
  const bool hi  = lane >= 16;
  const int m0 = blockIdx.y * 64  + (wave & 1) * 32;
  const int n0 = blockIdx.x * 128 + (wave >> 1) * 32;

  const bf16_t* a0 = A  + (size_t)m0 * K;
  const bf16_t* a1 = A  + (size_t)(m0 + 16) * K;
  const bf16_t* b0 = Wt + (size_t)n0 * K;
  const bf16_t* b1 = Wt + (size_t)(n0 + 16) * K;

  v8f acc[2][2];
  acc[0][0] = zero8(); acc[0][1] = zero8();
  acc[1][0] = zero8(); acc[1][1] = zero8();

#pragma unroll 2
  for (int kk = 0; kk < K; kk += 32) {
    const v16bf af0 = load_a(a0, K, kk, lane);
    const v16bf af1 = load_a(a1, K, kk, lane);
    const v16bf bf0 = load_b_cm(b0, K, kk, lane);
    const v16bf bf1 = load_b_cm(b1, K, kk, lane);
    acc[0][0] = wmma_bf16(af0, bf0, acc[0][0]);
    acc[0][1] = wmma_bf16(af0, bf1, acc[0][1]);
    acc[1][0] = wmma_bf16(af1, bf0, acc[1][0]);
    acc[1][1] = wmma_bf16(af1, bf1, acc[1][1]);
  }

#pragma unroll
  for (int mi = 0; mi < 2; ++mi) {
#pragma unroll
    for (int ni = 0; ni < 2; ++ni) {
      const int col = n0 + ni * 16 + l15;
      const float bb = bias[col];
#pragma unroll
      for (int r = 0; r < 8; ++r) {
        const int row = m0 + mi * 16 + r + (hi ? 8 : 0);
        const float v = acc[mi][ni][r] + bb;
        if (outf) outf[(size_t)row * N + col] = v;
        if (outh) outh[(size_t)row * N + col] = (bf16_t)v;
      }
    }
  }
}

// ---------------------------------------------------------------- flash attention
// Grid: (SEQ/64, NH, BATCH). Block: 128 threads (4 waves); wave owns 16 queries.
__global__ __launch_bounds__(128)
void mqa_attn_kernel(const bf16_t* __restrict__ Qh, const bf16_t* __restrict__ Kh,
                     const bf16_t* __restrict__ Vh, bf16_t* __restrict__ Ah) {
  __shared__ bf16_t Klds[64 * 64];       // [key][d]
  __shared__ bf16_t Vt[64 * 64];         // [d][key] (transposed)
  __shared__ bf16_t Plds[4][16 * 64];    // per-wave P scratch [row][key]

  const int lane = threadIdx.x & 31;
  const int wave = threadIdx.x >> 5;
  const int l15  = lane & 15;
  const bool hi  = lane >= 16;
  const int h = blockIdx.y;
  const int b = blockIdx.z;
  const int q0 = blockIdx.x * 64 + wave * 16;

  const bf16_t* qbase = Qh + ((size_t)(b * SEQ + q0)) * EMB + h * HD;
  const v16bf qa0 = load_a(qbase, EMB, 0,  lane);
  const v16bf qa1 = load_a(qbase, EMB, 32, lane);

  const bf16_t* kbase = Kh + (size_t)b * SEQ * HD;
  const bf16_t* vbase = Vh + (size_t)b * SEQ * HD;

  float m[8], l[8];
  v8f O[4];
#pragma unroll
  for (int r = 0; r < 8; ++r) { m[r] = -3.0e38f; l[r] = 0.0f; }
#pragma unroll
  for (int ct = 0; ct < 4; ++ct) O[ct] = zero8();

  for (int kb = 0; kb < SEQ; kb += 64) {
    __syncthreads();
    // K tile: async copy global -> LDS (ASYNCcnt path, no VGPR staging).
    {
      const bf16_t* gsrc = kbase + (size_t)kb * HD;
#pragma unroll
      for (int j = 0; j < 4; ++j) {
        const int c = threadIdx.x + j * 128;           // 512 x 16B chunks
        const unsigned ldsoff = (unsigned)(uintptr_t)(const void*)&Klds[c * 8];
        const bf16_t* g = gsrc + c * 8;
        asm volatile("global_load_async_to_lds_b128 %0, %1, off"
                     :: "v"(ldsoff), "v"(g) : "memory");
      }
    }
    // V tile: transpose through VGPRs (vector global loads, scalar LDS stores).
#pragma unroll
    for (int j = 0; j < 4; ++j) {
      const int c = threadIdx.x + j * 128;             // 512 chunks of 8
      const int ky = c >> 3, dd0 = (c & 7) * 8;
      const v8bf v = *(const v8bf*)(vbase + (size_t)(kb + ky) * HD + dd0);
#pragma unroll
      for (int e = 0; e < 8; ++e) Vt[(dd0 + e) * 64 + ky] = v[e];
    }
    asm volatile("s_wait_asynccnt 0" ::: "memory");
    __syncthreads();

    // scores: 16 queries x 64 keys, fp32 accumulate, scaled by 1/sqrt(64)
    v8f sc[4];
#pragma unroll
    for (int nt = 0; nt < 4; ++nt) {
      v8f a = zero8();
      a = wmma_bf16(qa0, load_b_cm(Klds + nt * 16 * 64, 64, 0,  lane), a);
      a = wmma_bf16(qa1, load_b_cm(Klds + nt * 16 * 64, 64, 32, lane), a);
      sc[nt] = a * 0.125f;
    }

    // online softmax (rows r / r+8 split by lane half, matching D layout)
#pragma unroll
    for (int r = 0; r < 8; ++r) {
      float mx = fmaxf(fmaxf(sc[0][r], sc[1][r]), fmaxf(sc[2][r], sc[3][r]));
#pragma unroll
      for (int msk = 8; msk >= 1; msk >>= 1) mx = fmaxf(mx, __shfl_xor(mx, msk, 32));
      const float mn = fmaxf(m[r], mx);
      const float alpha = __expf(m[r] - mn);
      m[r] = mn;
      float rs = 0.0f;
#pragma unroll
      for (int nt = 0; nt < 4; ++nt) {
        const float p = __expf(sc[nt][r] - mn);
        sc[nt][r] = p;
        rs += p;
      }
#pragma unroll
      for (int msk = 8; msk >= 1; msk >>= 1) rs += __shfl_xor(rs, msk, 32);
      l[r] = l[r] * alpha + rs;
#pragma unroll
      for (int ct = 0; ct < 4; ++ct) O[ct][r] *= alpha;
    }

    // D-layout -> A-layout via per-wave LDS scratch (bf16 P).
    bf16_t* Pw = Plds[wave];
#pragma unroll
    for (int nt = 0; nt < 4; ++nt)
#pragma unroll
      for (int r = 0; r < 8; ++r)
        Pw[(r + (hi ? 8 : 0)) * 64 + nt * 16 + l15] = (bf16_t)sc[nt][r];

    const v16bf p0 = load_a(Pw, 64, 0,  lane);
    const v16bf p1 = load_a(Pw, 64, 32, lane);
#pragma unroll
    for (int ct = 0; ct < 4; ++ct) {
      O[ct] = wmma_bf16(p0, load_b_cm(Vt + ct * 16 * 64, 64, 0,  lane), O[ct]);
      O[ct] = wmma_bf16(p1, load_b_cm(Vt + ct * 16 * 64, 64, 32, lane), O[ct]);
    }
  }

  // epilogue: normalize, store to [B][H][S][HD] (reference's faithful reshape)
#pragma unroll
  for (int r = 0; r < 8; ++r) {
    const float inv = 1.0f / l[r];
    const int row = q0 + r + (hi ? 8 : 0);
#pragma unroll
    for (int ct = 0; ct < 4; ++ct) {
      const int dd = ct * 16 + l15;
      Ah[(((size_t)b * NH + h) * SEQ + row) * HD + dd] = (bf16_t)(O[ct][r] * inv);
    }
  }
}

// ---------------------------------------------------------------- launch
extern "C" void kernel_launch(void* const* d_in, const int* in_sizes, int n_in,
                              void* d_out, int out_size, void* d_ws, size_t ws_size,
                              hipStream_t stream) {
  (void)in_sizes; (void)n_in; (void)out_size; (void)ws_size;
  const float* x  = (const float*)d_in[0];
  const float* wq = (const float*)d_in[1];
  const float* bq = (const float*)d_in[2];
  const float* wk = (const float*)d_in[3];
  const float* bk = (const float*)d_in[4];
  const float* wv = (const float*)d_in[5];
  const float* bv = (const float*)d_in[6];
  const float* wo = (const float*)d_in[7];
  const float* bo = (const float*)d_in[8];

  const int M = BATCH * SEQ;  // 4096
  char* ws = (char*)d_ws;
  size_t off = 0;
  auto alloc = [&](size_t bytes) {
    void* p = ws + off;
    off += (bytes + 255) & ~(size_t)255;
    return p;
  };
  bf16_t* xh  = (bf16_t*)alloc((size_t)M * EMB * 2);
  bf16_t* wqt = (bf16_t*)alloc((size_t)EMB * EMB * 2);
  bf16_t* wkt = (bf16_t*)alloc((size_t)HD * EMB * 2);
  bf16_t* wvt = (bf16_t*)alloc((size_t)HD * EMB * 2);
  bf16_t* wot = (bf16_t*)alloc((size_t)EMB * EMB * 2);
  bf16_t* Qh  = (bf16_t*)alloc((size_t)M * EMB * 2);
  bf16_t* Kh  = (bf16_t*)alloc((size_t)M * HD * 2);
  bf16_t* Vh  = (bf16_t*)alloc((size_t)M * HD * 2);
  bf16_t* Ah  = (bf16_t*)alloc((size_t)M * EMB * 2);

  const int nx = M * EMB;
  cvt_f32_bf16<<<(nx + 255) / 256, 256, 0, stream>>>(x, xh, nx);
  transpose_f32_bf16<<<(EMB * EMB + 255) / 256, 256, 0, stream>>>(wq, wqt, EMB, EMB);
  transpose_f32_bf16<<<(EMB * HD + 255) / 256, 256, 0, stream>>>(wk, wkt, EMB, HD);
  transpose_f32_bf16<<<(EMB * HD + 255) / 256, 256, 0, stream>>>(wv, wvt, EMB, HD);
  transpose_f32_bf16<<<(EMB * EMB + 255) / 256, 256, 0, stream>>>(wo, wot, EMB, EMB);

  // projections
  gemm_bf16_32x32<<<dim3(EMB / 128, M / 64), 256, 0, stream>>>(xh, wqt, bq, nullptr, Qh, M, EMB, EMB);
  gemm_bf16_16x16<<<dim3(1,         M / 32), 256, 0, stream>>>(xh, wkt, bk, nullptr, Kh, M, HD,  EMB);
  gemm_bf16_16x16<<<dim3(1,         M / 32), 256, 0, stream>>>(xh, wvt, bv, nullptr, Vh, M, HD,  EMB);

  // flash attention
  mqa_attn_kernel<<<dim3(SEQ / 64, NH, BATCH), 128, 0, stream>>>(Qh, Kh, Vh, Ah);

  // output projection (Ah interpreted as [M][E] == faithful reshape)
  gemm_bf16_32x32<<<dim3(EMB / 128, M / 64), 256, 0, stream>>>(Ah, wot, bo, (float*)d_out, nullptr, M, EMB, EMB);
}
